// spatial_model_67061619359900
// MI455X (gfx1250) — compile-verified, matrix operand order checked by reading, not run on hardware
//
#include <hip/hip_runtime.h>
#include <math.h>

#define Bn 512
#define Vn 32
#define Tn 20
#define Pn 10
#define Fn 2
#define Hn 256
#define G4 1024   // 4*H

typedef __bf16 bf16_t;
typedef bf16_t v16bf __attribute__((ext_vector_type(16)));
typedef float  v8f   __attribute__((ext_vector_type(8)));

union FragAB { v16bf v; uint4 q[2]; };

__device__ __forceinline__ float sigmoidf(float x){ return 1.0f/(1.0f + __expf(-x)); }
__device__ __forceinline__ float posmod360(float x){ float m = fmodf(x, 360.0f); return (m < 0.0f) ? m + 360.0f : m; }

__device__ __forceinline__ float attn_w(float d, float br, float hd, const float* __restrict__ dom,
                                        float mi, float mj){
    int ib = (int)floorf(br * (1.0f/30.0f)); ib = ib < 0 ? 0 : (ib > 11 ? 11 : ib);
    int ih = (int)floorf(hd * (1.0f/30.0f)); ih = ih < 0 ? 0 : (ih > 11 ? 11 : ih);
    return fmaxf(dom[ib*12 + ih] - d, 0.0f) * mi * mj;
}

// ---------------- weight fp32 -> bf16 conversion ----------------
__global__ __launch_bounds__(256) void cvt_bf16(const float* __restrict__ src,
                                                bf16_t* __restrict__ dst, int n){
    int i = blockIdx.x * 256 + threadIdx.x;
    if (i < n) dst[i] = (bf16_t)src[i];
}

// ---------------- prev = sequence[:,:,T-1,:] ----------------
__global__ __launch_bounds__(256) void copy_prev(const float* __restrict__ seq,
                                                 float* __restrict__ prev){
    int i = blockIdx.x * 256 + threadIdx.x;
    if (i < Bn*Vn*Fn){
        int row = i >> 1, j = i & 1;
        prev[i] = seq[((size_t)row * Tn + (Tn-1)) * Fn + j];
    }
}

// ---------------- encoder spatial attention: wh(bf16) = norm(w) @ h ----------------
__global__ __launch_bounds__(256) void attn_enc(const float* __restrict__ h,
                                                const float* __restrict__ dist,
                                                const float* __restrict__ bear,
                                                const float* __restrict__ head,
                                                const unsigned char* __restrict__ smask,
                                                const float* __restrict__ domain,
                                                bf16_t* __restrict__ whbf, int t){
    __shared__ float wn[Vn][Vn+1];
    __shared__ float msk[Vn];
    __shared__ float rinv[Vn];
    const int b = blockIdx.x, tid = threadIdx.x;

    if (tid < Vn) msk[tid] = smask[((size_t)(b*Vn + tid))*Tn + t] ? 1.0f : 0.0f;
    __syncthreads();

    for (int e = tid; e < Vn*Vn; e += 256){
        int i = e >> 5, j = e & 31;
        size_t base = (((size_t)(b*Vn + i))*Tn + t)*Vn + j;
        wn[i][j] = attn_w(dist[base], bear[base], head[base], domain, msk[i], msk[j]);
    }
    __syncthreads();
    if (tid < Vn){
        float s = 0.0f;
        for (int j = 0; j < Vn; ++j) s += wn[tid][j];
        rinv[tid] = 1.0f / (s + 1e-6f);
    }
    __syncthreads();

    const int hc = tid;                         // 0..255
    float hv[Vn];
    for (int j = 0; j < Vn; ++j) hv[j] = h[((size_t)(b*Vn + j))*Hn + hc];
    for (int i = 0; i < Vn; ++i){
        float acc = 0.0f;
        for (int j = 0; j < Vn; ++j) acc = fmaf(wn[i][j], hv[j], acc);
        whbf[((size_t)(b*Vn + i))*Hn + hc] = (bf16_t)(acc * rinv[i]);
    }
}

// ---------------- decoder spatial attention (features from pred/prev) ----------------
__global__ __launch_bounds__(256) void attn_dec(const float* __restrict__ h,
                                                const float* __restrict__ pred,
                                                const float* __restrict__ prev,
                                                const unsigned char* __restrict__ smask,
                                                const float* __restrict__ domain,
                                                bf16_t* __restrict__ whbf){
    __shared__ float wn[Vn][Vn+1];
    __shared__ float px[Vn], py[Vn], hdg[Vn], msk[Vn], rinv[Vn];
    const int b = blockIdx.x, tid = threadIdx.x;
    const float R2D = 57.29577951308232f;

    if (tid < Vn){
        int row = b*Vn + tid;
        float x = pred[row*2], y = pred[row*2+1];
        px[tid] = x; py[tid] = y;
        float mvx = x - prev[row*2], mvy = y - prev[row*2+1];
        hdg[tid]  = posmod360(atan2f(mvy, mvx + 1e-6f) * R2D);
        msk[tid]  = smask[((size_t)row)*Tn + (Tn-1)] ? 1.0f : 0.0f;
    }
    __syncthreads();

    for (int e = tid; e < Vn*Vn; e += 256){
        int i = e >> 5, j = e & 31;
        float rx = px[j] - px[i], ry = py[j] - py[i];
        float d  = sqrtf(rx*rx + ry*ry + 1e-12f);
        float eye = (i == j) ? 1e-6f : 0.0f;
        float ang = posmod360(atan2f(ry, rx + eye) * R2D);
        float br  = posmod360(ang - hdg[i]);
        float hd  = posmod360(hdg[j] - hdg[i]);
        wn[i][j] = attn_w(d, br, hd, domain, msk[i], msk[j]);
    }
    __syncthreads();
    if (tid < Vn){
        float s = 0.0f;
        for (int j = 0; j < Vn; ++j) s += wn[tid][j];
        rinv[tid] = 1.0f / (s + 1e-6f);
    }
    __syncthreads();

    const int hc = tid;
    float hv[Vn];
    for (int j = 0; j < Vn; ++j) hv[j] = h[((size_t)(b*Vn + j))*Hn + hc];
    for (int i = 0; i < Vn; ++i){
        float acc = 0.0f;
        for (int j = 0; j < Vn; ++j) acc = fmaf(wn[i][j], hv[j], acc);
        whbf[((size_t)(b*Vn + i))*Hn + hc] = (bf16_t)(acc * rinv[i]);
    }
}

// ---------------- fused LSTM step: WMMA GEMM + gates (+ decoder pred) ----------------
__global__ __launch_bounds__(256) void lstm_step(
    const bf16_t* __restrict__ Abf,   // (B*V, H) bf16 attention-mixed hidden
    const bf16_t* __restrict__ Wbf,   // (4H, H)  bf16 W_hh
    const float*  __restrict__ Wih,   // (4H, 2)
    const float*  __restrict__ bih,
    const float*  __restrict__ bhh,
    const float*  __restrict__ xsrc,  // sequence (B,V,T,F) or prev (B*V,2)
    int x_is_seq, int xt,
    const unsigned char* __restrict__ smask, int mask_t,
    float* __restrict__ hbuf,
    float* __restrict__ cbuf,
    const float* __restrict__ Wout,   // (2,H) or null
    const float* __restrict__ bout,
    float* __restrict__ pred_buf,     // (B*V,2) or null
    float* __restrict__ out,          // d_out (B,V,P,F) or null
    int pstep)
{
    // 16 rows x 1024 gate cols, padded stride 1028 (== 4 mod 64 banks)
    __shared__ float g_s[16][1028];
    const int tid  = threadIdx.x;
    const int wave = tid >> 5, lane = tid & 31;
    const int mb   = blockIdx.x;
    const int half = lane >> 4, lr = lane & 15;

    // ---- Phase 1: g = A @ W^T via v_wmma_f32_16x16x32_bf16 ----
    // 16-bit A layout: lane<16 holds K {0..7,16..23}+k, lane>=16 holds K {8..15,24..31}+k
    const bf16_t* arow = Abf + ((size_t)(mb*16 + lr))*Hn + half*8;
    const int n_base = wave * 128;
    v8f acc[8] = {};
    for (int k = 0; k < Hn; k += 32){
        FragAB a;
        a.q[0] = *(const uint4*)(arow + k);
        a.q[1] = *(const uint4*)(arow + k + 16);
#pragma unroll
        for (int j = 0; j < 8; ++j){
            const bf16_t* brow = Wbf + ((size_t)(n_base + j*16 + lr))*Hn + half*8;
            FragAB bb;
            bb.q[0] = *(const uint4*)(brow + k);
            bb.q[1] = *(const uint4*)(brow + k + 16);
            acc[j] = __builtin_amdgcn_wmma_f32_16x16x32_bf16(
                         false, a.v, false, bb.v, (short)0, acc[j], false, false);
        }
    }
    // D layout: VGPR r -> M = r + 8*(lane/16), N = n0 + lane%16
#pragma unroll
    for (int j = 0; j < 8; ++j){
        const int n0 = n_base + j*16 + lr;
        const int m0 = half*8;
#pragma unroll
        for (int r = 0; r < 8; ++r) g_s[m0 + r][n0] = acc[j][r];
    }
    __syncthreads();

    // ---- Phase 2: x@W_ih + biases, gates, cell update ----
    const int hc = tid;                                  // 0..255 (= H)
    const float wi0 = Wih[(hc       )*2], wi1 = Wih[(hc       )*2+1], bi = bih[hc      ] + bhh[hc      ];
    const float wf0 = Wih[(hc + 256 )*2], wf1 = Wih[(hc + 256 )*2+1], bf = bih[hc + 256] + bhh[hc + 256];
    const float wg0 = Wih[(hc + 512 )*2], wg1 = Wih[(hc + 512 )*2+1], bg = bih[hc + 512] + bhh[hc + 512];
    const float wo0 = Wih[(hc + 768 )*2], wo1 = Wih[(hc + 768 )*2+1], bo = bih[hc + 768] + bhh[hc + 768];

    for (int m = 0; m < 16; ++m){
        const int row = mb*16 + m;
        float x0, x1;
        if (x_is_seq){
            size_t xi = ((size_t)row * Tn + xt) * Fn;
            x0 = xsrc[xi]; x1 = xsrc[xi + 1];
        } else {
            x0 = xsrc[row*2]; x1 = xsrc[row*2 + 1];
        }
        float gi = g_s[m][hc      ] + x0*wi0 + x1*wi1 + bi;
        float gf = g_s[m][hc + 256] + x0*wf0 + x1*wf1 + bf;
        float gg = g_s[m][hc + 512] + x0*wg0 + x1*wg1 + bg;
        float go = g_s[m][hc + 768] + x0*wo0 + x1*wo1 + bo;

        float c  = cbuf[(size_t)row*Hn + hc];
        float c2 = sigmoidf(gf)*c + sigmoidf(gi)*tanhf(gg);
        float mv = smask[(size_t)row*Tn + mask_t] ? 1.0f : 0.0f;
        float h2 = sigmoidf(go)*tanhf(c2) * mv;

        cbuf[(size_t)row*Hn + hc] = c2;
        hbuf[(size_t)row*Hn + hc] = h2;
        if (pred_buf) g_s[m][hc] = h2;   // stash for output-projection reduction
    }

    // ---- decoder only: pred = relu(h2 @ Wout^T + bout) * mask ----
    if (pred_buf){
        __syncthreads();
        if (tid < 32){
            const int m = tid >> 1, j = tid & 1;
            float acc2 = bout[j];
            for (int k = 0; k < Hn; ++k) acc2 = fmaf(g_s[m][k], Wout[j*Hn + k], acc2);
            const int row = mb*16 + m;
            float mv = smask[(size_t)row*Tn + mask_t] ? 1.0f : 0.0f;
            float p  = fmaxf(acc2, 0.0f) * mv;
            pred_buf[row*2 + j] = p;
            out[((size_t)row * Pn + pstep)*Fn + j] = p;
        }
    }
}

extern "C" void kernel_launch(void* const* d_in, const int* in_sizes, int n_in,
                              void* d_out, int out_size, void* d_ws, size_t ws_size,
                              hipStream_t stream)
{
    const float* seq    = (const float*)d_in[0];
    const float* dist   = (const float*)d_in[1];
    const float* bear   = (const float*)d_in[2];
    const float* head   = (const float*)d_in[3];
    const unsigned char* smask = (const unsigned char*)d_in[4];
    /* d_in[5] op_mask: reference uses seq_mask[:,:,-1] instead */
    const float* domain = (const float*)d_in[6];
    const float* WihE   = (const float*)d_in[7];
    const float* WhhE   = (const float*)d_in[8];
    const float* bihE   = (const float*)d_in[9];
    const float* bhhE   = (const float*)d_in[10];
    const float* WihD   = (const float*)d_in[11];
    const float* WhhD   = (const float*)d_in[12];
    const float* bihD   = (const float*)d_in[13];
    const float* bhhD   = (const float*)d_in[14];
    const float* Wout   = (const float*)d_in[15];
    const float* bout   = (const float*)d_in[16];
    float* out = (float*)d_out;

    const size_t HN = (size_t)Bn * Vn * Hn;     // 16384*256
    char* w = (char*)d_ws;
    float*  hbuf = (float*)w;  w += HN * 4;
    float*  cbuf = (float*)w;  w += HN * 4;
    bf16_t* whbf = (bf16_t*)w; w += HN * 2;
    bf16_t* WbfE = (bf16_t*)w; w += (size_t)G4 * Hn * 2;
    bf16_t* WbfD = (bf16_t*)w; w += (size_t)G4 * Hn * 2;
    float* pb0 = (float*)w;    w += (size_t)Bn * Vn * Fn * 4;
    float* pb1 = (float*)w;    w += (size_t)Bn * Vn * Fn * 4;

    hipMemsetAsync(hbuf, 0, HN * 4, stream);
    hipMemsetAsync(cbuf, 0, HN * 4, stream);
    cvt_bf16<<<(G4*Hn + 255)/256, 256, 0, stream>>>(WhhE, WbfE, G4*Hn);
    cvt_bf16<<<(G4*Hn + 255)/256, 256, 0, stream>>>(WhhD, WbfD, G4*Hn);

    const int nblk = (Bn * Vn) / 16;            // 1024
    for (int t = 0; t < Tn; ++t){
        attn_enc<<<Bn, 256, 0, stream>>>(hbuf, dist, bear, head, smask, domain, whbf, t);
        lstm_step<<<nblk, 256, 0, stream>>>(whbf, WbfE, WihE, bihE, bhhE,
                                            seq, 1, t, smask, t, hbuf, cbuf,
                                            nullptr, nullptr, nullptr, nullptr, 0);
    }

    // decoder init: attention over final h with t = T-1 features, op_mask = seq_mask[:,:,-1]
    attn_enc<<<Bn, 256, 0, stream>>>(hbuf, dist, bear, head, smask, domain, whbf, Tn-1);
    copy_prev<<<(Bn*Vn*Fn + 255)/256, 256, 0, stream>>>(seq, pb0);

    for (int p = 0; p < Pn; ++p){
        float* prev = (p & 1) ? pb1 : pb0;
        float* pred = (p & 1) ? pb0 : pb1;
        lstm_step<<<nblk, 256, 0, stream>>>(whbf, WbfD, WihD, bihD, bhhD,
                                            prev, 0, 0, smask, Tn-1, hbuf, cbuf,
                                            Wout, bout, pred, out, p);
        if (p + 1 < Pn)
            attn_dec<<<Bn, 256, 0, stream>>>(hbuf, pred, prev, smask, domain, whbf);
    }
}
